// ISTFTHead_43009802502778
// MI455X (gfx1250) — matrix operand order repaired and checked by reading, not compiled, not run on hardware
//
#include <hip/hip_runtime.h>

typedef __attribute__((ext_vector_type(16))) __bf16 v16bf;
typedef __attribute__((ext_vector_type(8)))  float  v8f;

#define BATCH 16
#define DIMC  512
#define LFRM  4096
#define NFFT  1024
#define HOP   256
#define WINL  1024
#define KP    528            // 513 bins padded to 33 tiles of 16
#define K2    (2*KP)         // 1056 interleaved re/im rows
#define OUTT  1048576        // trimmed output length per batch
#define PADT  384

#define TWO_PI 6.28318530717958647692f

__device__ __forceinline__ unsigned short f2bf(float f) {
    unsigned u = __float_as_uint(f);
    unsigned r = 0x7FFFu + ((u >> 16) & 1u);   // round-to-nearest-even
    return (unsigned short)((u + r) >> 16);
}
__device__ __forceinline__ float bf2f(unsigned short h) {
    return __uint_as_float(((unsigned)h) << 16);
}

union Frag {
    v16bf v;
    unsigned short s[16];
    uint4 q[2];
};

// ---------------- init: split W into bf16 hi/lo planes ----------------
__global__ void wsplit_kernel(const float* __restrict__ W,
                              unsigned short* __restrict__ Whi,
                              unsigned short* __restrict__ Wlo, int n) {
    int i = blockIdx.x * blockDim.x + threadIdx.x;
    if (i >= n) return;
    float f = W[i];
    unsigned short hi = f2bf(f);
    Whi[i] = hi;
    Wlo[i] = f2bf(f - bf2f(hi));
}

// ---------------- init: windowed IDFT basis (1024 x 1056), bf16 hi/lo ----------------
__global__ void basis_kernel(unsigned short* __restrict__ bHi,
                             unsigned short* __restrict__ bLo) {
    int i = blockIdx.x * blockDim.x + threadIdx.x;
    if (i >= NFFT * K2) return;
    int nrow = i / K2;
    int k2   = i - nrow * K2;
    int k    = k2 >> 1;
    float val = 0.f;
    if (k <= 512) {
        float scale = (k == 0 || k == 512) ? 1.f : 2.f;
        float ang = TWO_PI * (float)k * (float)nrow / (float)NFFT;
        float win = 0.5f * (1.f - cosf(TWO_PI * (float)nrow / (float)WINL));
        float base = (k2 & 1) ? -sinf(ang) : cosf(ang);
        val = scale * base * win * (1.f / (float)NFFT);
    }
    unsigned short hi = f2bf(val);
    bHi[i] = hi;
    bLo[i] = f2bf(val - bf2f(hi));
}

// ---------------- zero the (poisoned) output ----------------
__global__ void zero_kernel(float* __restrict__ p, int n) {
    int i = blockIdx.x * blockDim.x + threadIdx.x;
    if (i < n) p[i] = 0.f;
}

// ---------------- per-batch: transpose x (DIM,L) -> bf16 hi/lo planes (L,DIM) ----------------
__global__ void xsplit_kernel(const float* __restrict__ xb,          // (DIMC, LFRM)
                              unsigned short* __restrict__ xTh,      // (LFRM, DIMC)
                              unsigned short* __restrict__ xTl)
{
    __shared__ float tile[32][33];
    const int l0 = blockIdx.x * 32;
    const int d0 = blockIdx.y * 32;
    const int tx = threadIdx.x & 31;
    const int ty = threadIdx.x >> 5;       // 0..7
    #pragma unroll
    for (int r = 0; r < 32; r += 8)
        tile[ty + r][tx] = xb[(size_t)(d0 + ty + r) * LFRM + (l0 + tx)];
    __syncthreads();
    #pragma unroll
    for (int r = 0; r < 32; r += 8) {
        float f = tile[tx][ty + r];
        unsigned short hi = f2bf(f);
        size_t o = (size_t)(l0 + ty + r) * DIMC + (d0 + tx);
        xTh[o] = hi;
        xTl[o] = f2bf(f - bf2f(hi));
    }
}

// ---------------- GEMM1 (W @ x + b) fused with exp/clip/cos/sin ----------------
// Per wave: one 16x16 mag tile + the paired phase tile (rows +NFFT), bf16x3.
// All fragments are contiguous uint4 loads; epilogue writes transposed
// hi/lo S planes (L, K2) with re/im interleaved along k2 (GEMM2's B layout).
__global__ void __launch_bounds__(256, 2)
gemm1_act_kernel(const unsigned short* __restrict__ xTh,  // (LFRM, DIMC)
                 const unsigned short* __restrict__ xTl,
                 const unsigned short* __restrict__ Whi,  // (2*NFFT, DIMC)
                 const unsigned short* __restrict__ Wlo,
                 const float* __restrict__ bias,          // (2*NFFT,)
                 unsigned short* __restrict__ S2h,        // (LFRM, K2)
                 unsigned short* __restrict__ S2l)
{
    const int lane    = threadIdx.x & 31;
    const int wave    = threadIdx.x >> 5;
    const int coltile = blockIdx.x * 8 + wave;   // 0..255
    const int otile   = blockIdx.y;              // 0..32
    const int l0 = coltile * 16;
    const int o0 = otile * 16;
    const int nn = lane & 15;
    const int hh = lane >> 4;
    const int l  = l0 + nn;

    // A rows (ISA 16-bit A layout: M = lane&15)
    const int mrow = o0 + nn;
    const unsigned short* WhiM = Whi + (size_t)mrow * DIMC;
    const unsigned short* WloM = Wlo + (size_t)mrow * DIMC;
    const unsigned short* WhiP = Whi + (size_t)(NFFT + mrow) * DIMC;
    const unsigned short* WloP = Wlo + (size_t)(NFFT + mrow) * DIMC;
    // B column for this lane (N = lane&15; this code's K order: K = 16*(lane>>4)+i, same as A)
    const unsigned short* xhR = xTh + (size_t)l * DIMC;
    const unsigned short* xlR = xTl + (size_t)l * DIMC;

    v8f accM = {};
    v8f accP = {};

    for (int ks = 0; ks < DIMC / 32; ++ks) {
        const int kb  = ks * 32;
        const int ka  = kb + 16 * hh;   // A: 16 contiguous K values
        const int kbh = kb + 16 * hh;   // B: 16 contiguous K values

        Frag bh, bl;
        bh.q[0] = *(const uint4*)(xhR + kbh);
        bh.q[1] = *(const uint4*)(xhR + kbh + 8);
        bl.q[0] = *(const uint4*)(xlR + kbh);
        bl.q[1] = *(const uint4*)(xlR + kbh + 8);

        Frag amh, aml, aph, apl;
        amh.q[0] = *(const uint4*)(WhiM + ka);
        amh.q[1] = *(const uint4*)(WhiM + ka + 8);
        aml.q[0] = *(const uint4*)(WloM + ka);
        aml.q[1] = *(const uint4*)(WloM + ka + 8);
        aph.q[0] = *(const uint4*)(WhiP + ka);
        aph.q[1] = *(const uint4*)(WhiP + ka + 8);
        apl.q[0] = *(const uint4*)(WloP + ka);
        apl.q[1] = *(const uint4*)(WloP + ka + 8);

        // bf16x3: hh + hl + lh
        accM = __builtin_amdgcn_wmma_f32_16x16x32_bf16(false, amh.v, false, bh.v, (short)0, accM, false, false);
        accM = __builtin_amdgcn_wmma_f32_16x16x32_bf16(false, amh.v, false, bl.v, (short)0, accM, false, false);
        accM = __builtin_amdgcn_wmma_f32_16x16x32_bf16(false, aml.v, false, bh.v, (short)0, accM, false, false);
        accP = __builtin_amdgcn_wmma_f32_16x16x32_bf16(false, aph.v, false, bh.v, (short)0, accP, false, false);
        accP = __builtin_amdgcn_wmma_f32_16x16x32_bf16(false, aph.v, false, bl.v, (short)0, accP, false, false);
        accP = __builtin_amdgcn_wmma_f32_16x16x32_bf16(false, apl.v, false, bh.v, (short)0, accP, false, false);
    }

    // C layout: row = r + 8*hh, col = nn. This lane owns contiguous k2 range
    // [2*(o0+8*hh), +16) in the transposed S planes -> 2 uint4 stores per plane.
    Frag outh, outl;
    #pragma unroll
    for (int r = 0; r < 8; ++r) {
        int row = o0 + r + 8 * hh;
        float mag = accM[r] + bias[row];
        mag = fminf(__expf(mag), 100.f);
        float ph = accP[r] + bias[NFFT + row];
        float re = mag * __cosf(ph);
        float im = mag * __sinf(ph);
        unsigned short rh = f2bf(re);
        unsigned short ih = f2bf(im);
        outh.s[2 * r]     = rh;
        outh.s[2 * r + 1] = ih;
        outl.s[2 * r]     = f2bf(re - bf2f(rh));
        outl.s[2 * r + 1] = f2bf(im - bf2f(ih));
    }
    const int k2s = 2 * (o0 + 8 * hh);
    unsigned short* dh = S2h + (size_t)l * K2 + k2s;
    unsigned short* dl = S2l + (size_t)l * K2 + k2s;
    *(uint4*)(dh)     = outh.q[0];
    *(uint4*)(dh + 8) = outh.q[1];
    *(uint4*)(dl)     = outl.q[0];
    *(uint4*)(dl + 8) = outl.q[1];
}

// ---------------- GEMM2 (windowed IDFT) fused with overlap-add scatter ----------------
__global__ void __launch_bounds__(256, 2)
gemm2_ola_kernel(const unsigned short* __restrict__ S2h,   // (LFRM, K2)
                 const unsigned short* __restrict__ S2l,
                 const unsigned short* __restrict__ bHi,   // (NFFT, K2)
                 const unsigned short* __restrict__ bLo,
                 float* __restrict__ outb)                 // (OUTT,)
{
    const int lane    = threadIdx.x & 31;
    const int wave    = threadIdx.x >> 5;
    const int coltile = blockIdx.x * 8 + wave;   // 0..255
    const int ntile   = blockIdx.y;              // 0..63
    const int l0 = coltile * 16;
    const int n0 = ntile * 16;
    const int nn = lane & 15;
    const int hh = lane >> 4;
    const int l  = l0 + nn;

    const unsigned short* bHiR = bHi + (size_t)(n0 + nn) * K2;
    const unsigned short* bLoR = bLo + (size_t)(n0 + nn) * K2;
    const unsigned short* shR  = S2h + (size_t)l * K2;
    const unsigned short* slR  = S2l + (size_t)l * K2;

    v8f acc = {};

    for (int ks = 0; ks < K2 / 32; ++ks) {
        const int kb  = ks * 32;
        const int ka  = kb + 16 * hh;
        const int kbh = kb + 16 * hh;

        Frag bhf, blf;
        bhf.q[0] = *(const uint4*)(shR + kbh);
        bhf.q[1] = *(const uint4*)(shR + kbh + 8);
        blf.q[0] = *(const uint4*)(slR + kbh);
        blf.q[1] = *(const uint4*)(slR + kbh + 8);

        Frag ah, al;
        ah.q[0] = *(const uint4*)(bHiR + ka);
        ah.q[1] = *(const uint4*)(bHiR + ka + 8);
        al.q[0] = *(const uint4*)(bLoR + ka);
        al.q[1] = *(const uint4*)(bLoR + ka + 8);

        acc = __builtin_amdgcn_wmma_f32_16x16x32_bf16(false, ah.v, false, bhf.v, (short)0, acc, false, false);
        acc = __builtin_amdgcn_wmma_f32_16x16x32_bf16(false, ah.v, false, blf.v, (short)0, acc, false, false);
        acc = __builtin_amdgcn_wmma_f32_16x16x32_bf16(false, al.v, false, bhf.v, (short)0, acc, false, false);
    }

    // overlap-add: sample index t = l*HOP + nrow - PADT into trimmed output
    #pragma unroll
    for (int r = 0; r < 8; ++r) {
        int nrow = n0 + r + 8 * hh;
        int t = l * HOP + nrow - PADT;
        if (t >= 0 && t < OUTT)
            atomicAdd(outb + t, acc[r]);
    }
}

// ---------------- envelope normalization (analytic Hann^2 sum) ----------------
__global__ void finalize_kernel(float* __restrict__ out, int total) {
    int i = blockIdx.x * blockDim.x + threadIdx.x;
    if (i >= total) return;
    int t  = i % OUTT;
    int tf = t + PADT;
    int jmax = tf / HOP;
    if (jmax > LFRM - 1) jmax = LFRM - 1;
    int jmin = jmax - 3;
    if (jmin < 0) jmin = 0;
    float env = 0.f;
    for (int j = jmin; j <= jmax; ++j) {
        int n = tf - j * HOP;
        if (n < WINL) {
            float w = 0.5f * (1.f - __cosf(TWO_PI * (float)n / (float)WINL));
            env += w * w;
        }
    }
    out[i] = out[i] / env;
}

extern "C" void kernel_launch(void* const* d_in, const int* in_sizes, int n_in,
                              void* d_out, int out_size, void* d_ws, size_t ws_size,
                              hipStream_t stream) {
    const float* x    = (const float*)d_in[0];   // (B, DIM, L)
    const float* W    = (const float*)d_in[1];   // (2*NFFT, DIM)
    const float* bias = (const float*)d_in[2];   // (2*NFFT,)
    float* out = (float*)d_out;                  // (B, OUTT)

    char* ws = (char*)d_ws;
    const size_t szW = (size_t)2 * NFFT * DIMC * sizeof(unsigned short); // 2 MB
    const size_t szB = (size_t)NFFT * K2 * sizeof(unsigned short);       // 2.1 MB
    const size_t szX = (size_t)LFRM * DIMC * sizeof(unsigned short);     // 4.2 MB
    const size_t szS = (size_t)LFRM * K2 * sizeof(unsigned short);       // 8.7 MB
    size_t off = 0;
    unsigned short* Whi = (unsigned short*)(ws + off); off += szW;
    unsigned short* Wlo = (unsigned short*)(ws + off); off += szW;
    unsigned short* bHi = (unsigned short*)(ws + off); off += szB;
    unsigned short* bLo = (unsigned short*)(ws + off); off += szB;
    unsigned short* xTh = (unsigned short*)(ws + off); off += szX;
    unsigned short* xTl = (unsigned short*)(ws + off); off += szX;
    unsigned short* S2h = (unsigned short*)(ws + off); off += szS;
    unsigned short* S2l = (unsigned short*)(ws + off); off += szS;

    const int nW = 2 * NFFT * DIMC;
    wsplit_kernel<<<(nW + 255) / 256, 256, 0, stream>>>(W, Whi, Wlo, nW);

    const int nB = NFFT * K2;
    basis_kernel<<<(nB + 255) / 256, 256, 0, stream>>>(bHi, bLo);

    const int nOut = BATCH * OUTT;
    zero_kernel<<<(nOut + 255) / 256, 256, 0, stream>>>(out, nOut);

    for (int b = 0; b < BATCH; ++b) {
        xsplit_kernel<<<dim3(LFRM / 32, DIMC / 32), 256, 0, stream>>>(
            x + (size_t)b * DIMC * LFRM, xTh, xTl);
        gemm1_act_kernel<<<dim3(32, 33), 256, 0, stream>>>(
            xTh, xTl, Whi, Wlo, bias, S2h, S2l);
        gemm2_ola_kernel<<<dim3(32, 64), 256, 0, stream>>>(
            S2h, S2l, bHi, bLo, out + (size_t)b * OUTT);
    }

    finalize_kernel<<<(nOut + 255) / 256, 256, 0, stream>>>(out, nOut);
}
